// Dense_01_56779467653690
// MI455X (gfx1250) — compile-verified, matrix operand order checked by reading, not run on hardware
//
#include <hip/hip_runtime.h>
#include <stdint.h>

// ---------------------------------------------------------------------------
// Binary-weight dense layer:  out[M,N] = scale * x[M,K] @ W[K,N], W in {+1,-1}
// M = B*S = 16384, K = 1024, N = 1024.
// bf16 WMMA (V_WMMA_F32_16X16X32_BF16), scale folded into packed weights.
// Workgroup: 256 threads (8 wave32) -> 128(M) x 256(N) tile, double-buffered
// LDS, async global->LDS (inline asm, ASYNCcnt) for the weight stream.
// ---------------------------------------------------------------------------

typedef __bf16 bf16_t;
typedef bf16_t v16bf __attribute__((ext_vector_type(16)));
typedef bf16_t v2bf  __attribute__((ext_vector_type(2)));
typedef float  v8f   __attribute__((ext_vector_type(8)));

#define M_TOT 16384u
#define N_TOT 1024u
#define K_TOT 1024u
#define KSTEPS (K_TOT / 32u)

#if defined(__gfx1250__)
#define ASYNC_LDS 1
#else
#define ASYNC_LDS 0
#endif

static __device__ __forceinline__ void wait_async_all() {
#if ASYNC_LDS
    asm volatile("s_wait_asynccnt 0x0" ::: "memory");
#endif
}

__device__ __forceinline__ unsigned f32_to_bf16_rne(float f) {
    union { float f; unsigned int u; } c;
    c.f = f;
    unsigned int u = c.u;
    unsigned int r = u + 0x7FFFu + ((u >> 16) & 1u);
    return r >> 16;   // low 16 bits = bf16
}

// Pack two fp32 -> packed bf16x2. Prefer the packed hardware convert builtin,
// else plain fptrunc casts (RNE) which the backend can select to the native
// bf16 convert on gfx1250 (and at worst expands to the integer sequence).
__device__ __forceinline__ unsigned pack2_bf16(float a, float b) {
#if __has_builtin(__builtin_amdgcn_cvt_pk_bf16_f32)
    union { v2bf v; unsigned u; } r;
    r.v = __builtin_amdgcn_cvt_pk_bf16_f32(a, b);
    return r.u;
#else
    union { v2bf v; unsigned u; } r;
    r.v[0] = (bf16_t)a;
    r.v[1] = (bf16_t)b;
    return r.u;
#endif
}

// ---------------------------------------------------------------------------
// Pack kernel: bool W[k][n] -> bf16 (+scale/-scale) in WMMA B-fragment order.
//   ushort index o = nt*16384 + kt*512 + eb*256 + lane*8 + j
//   lane = half*16 + n_local ; K_local = j + 8*half + 16*eb  (ISA 7.12.2)
// ---------------------------------------------------------------------------
__global__ __launch_bounds__(256) void bin_pack_w_kernel(
    const unsigned char* __restrict__ wbool,   // [K][N] bool (1 byte)
    const float* __restrict__ scale_p,
    unsigned short* __restrict__ wp)
{
    const unsigned o = blockIdx.x * 256u + threadIdx.x;   // 0 .. 2^20-1
    const float s = scale_p[0];

    const unsigned j    = o & 7u;
    const unsigned lane = (o >> 3) & 31u;
    const unsigned eb   = (o >> 8) & 1u;
    const unsigned kt   = (o >> 9) & 31u;
    const unsigned nt   = o >> 14;

    const unsigned half = lane >> 4;
    const unsigned nloc = lane & 15u;
    const unsigned kloc = j + 8u * half + 16u * eb;

    const unsigned k = kt * 32u + kloc;
    const unsigned n = nt * 16u + nloc;

    const float v = wbool[(size_t)k * N_TOT + n] ? s : -s;
    wp[o] = (unsigned short)f32_to_bf16_rne(v);
}

// ---------------------------------------------------------------------------
// GEMM kernel.
//   waves: 2(M) x 4(N); each wave owns 4x4 16x16 tiles (64x64 output block).
//   Per K-step(32): each wave = 16 ds_load_b128 + 16 v_wmma.
//   Double-buffered LDS; B streamed with GLOBAL_LOAD_ASYNC_TO_LDS_B128.
// ---------------------------------------------------------------------------
__global__ __launch_bounds__(256) void bin_dense_wmma_kernel(
    const float* __restrict__ x,               // [M][K] fp32
    const unsigned short* __restrict__ wp,     // packed bf16 weights
    float* __restrict__ out)                   // [M][N] fp32
{
    __shared__ unsigned short ldsA[2][8 * 512];    // 2 x 8 KB  : [sm][eb][lane][8]
    __shared__ unsigned short ldsB[2][16 * 512];   // 2 x 16 KB : [sn][eb][lane][8]

    const unsigned tid    = threadIdx.x;
    const unsigned lane   = tid & 31u;
    const unsigned wave   = tid >> 5;
    const unsigned wave_m = wave >> 2;    // 0..1
    const unsigned wave_n = wave & 3u;    // 0..3

    const unsigned nblk = blockIdx.x & 3u;          // 4 N blocks of 256
    const unsigned mblk = blockIdx.x >> 2;          // 128 M blocks of 128
    const unsigned m0   = mblk * 128u;
    const unsigned nt0  = nblk * 16u;               // base 16-col subtile index

    v8f acc[4][4];
    const v8f zero = {};
    #pragma unroll
    for (int i = 0; i < 4; ++i)
        #pragma unroll
        for (int j = 0; j < 4; ++j) acc[i][j] = zero;

    float4 areg[4];
#if !ASYNC_LDS
    uint4 breg[4];
#endif

    // ---- issue global loads for K-step kt; B goes async straight to LDS ----
    auto issue_stage = [&](unsigned kt, unsigned buf) {
        #pragma unroll
        for (int i = 0; i < 4; ++i) {
            const unsigned c    = tid + 256u * i;   // 0..1023: 4-float chunks
            const unsigned mloc = c >> 3;
            const unsigned kq   = c & 7u;
            areg[i] = *(const float4*)(x + (size_t)(m0 + mloc) * K_TOT
                                         + kt * 32u + kq * 4u);
        }
        #pragma unroll
        for (int i = 0; i < 4; ++i) {
            const unsigned c = tid + 256u * i;      // 0..1023: 16B chunks
            const unsigned short* src =
                wp + (size_t)((nt0 + (c >> 6)) * 32u + kt) * 512u + (c & 63u) * 8u;
#if ASYNC_LDS
            // Per-lane LDS byte offset: low 32 bits of the generic pointer to a
            // __shared__ object are the allocation-relative LDS offset.
            const unsigned lds_addr = (unsigned)(size_t)&ldsB[buf][c * 8u];
            asm volatile("global_load_async_to_lds_b128 %0, %1, off"
                         :: "v"(lds_addr), "v"(src)
                         : "memory");
#else
            breg[i] = *(const uint4*)src;
#endif
        }
    };

    // ---- convert/store staged registers into LDS ----
    auto store_stage = [&](unsigned buf) {
        #pragma unroll
        for (int i = 0; i < 4; ++i) {
            const unsigned c    = tid + 256u * i;
            const unsigned mloc = c >> 3;
            const unsigned kq   = c & 7u;
            uint2 p;
            p.x = pack2_bf16(areg[i].x, areg[i].y);
            p.y = pack2_bf16(areg[i].z, areg[i].w);
            const unsigned sm = mloc >> 4;          // 0..7
            const unsigned r  = mloc & 15u;
            const unsigned kl = kq * 4u;            // 0,4,...,28
            const unsigned eb = kl >> 4;
            const unsigned hf = (kl >> 3) & 1u;
            const unsigned jj = kl & 7u;
            *(uint2*)&ldsA[buf][sm * 512u + eb * 256u + (hf * 16u + r) * 8u + jj] = p;
        }
#if !ASYNC_LDS
        #pragma unroll
        for (int i = 0; i < 4; ++i) {
            const unsigned c = tid + 256u * i;
            *(uint4*)&ldsB[buf][c * 8u] = breg[i];
        }
#endif
    };

    // ---- 16 WMMAs on the current LDS buffer ----
    auto compute_step = [&](unsigned buf) {
        union frag_t { uint4 u[2]; v16bf v; };
        frag_t af[4];
        #pragma unroll
        for (int i = 0; i < 4; ++i) {
            const unsigned base = (wave_m * 4u + i) * 512u + lane * 8u;
            af[i].u[0] = *(const uint4*)&ldsA[buf][base];
            af[i].u[1] = *(const uint4*)&ldsA[buf][base + 256u];
        }
        #pragma unroll
        for (int j = 0; j < 4; ++j) {
            frag_t bfr;
            const unsigned base = (wave_n * 4u + j) * 512u + lane * 8u;
            bfr.u[0] = *(const uint4*)&ldsB[buf][base];
            bfr.u[1] = *(const uint4*)&ldsB[buf][base + 256u];
            #pragma unroll
            for (int i = 0; i < 4; ++i) {
                acc[i][j] = __builtin_amdgcn_wmma_f32_16x16x32_bf16(
                    false, af[i].v, false, bfr.v,
                    (short)0, acc[i][j], false, false);
            }
        }
    };

    // ---- prologue: fill buffer 0 ----
    issue_stage(0u, 0u);
    store_stage(0u);
    wait_async_all();
    __syncthreads();

    // ---- main loop: 1 barrier per K-step, staging overlapped with WMMA ----
    for (unsigned kt = 0; kt < KSTEPS; ++kt) {
        const unsigned buf = kt & 1u;
        if (kt + 1u < KSTEPS) issue_stage(kt + 1u, buf ^ 1u);
        compute_step(buf);
        if (kt + 1u < KSTEPS) {
            store_stage(buf ^ 1u);
            wait_async_all();
            __syncthreads();
        }
    }

    // ---- epilogue: C layout r -> M=r (lanes 0-15) / M=r+8 (lanes 16-31) ----
    // Non-temporal: out is streamed once, keep it out of L2 (which holds x).
    const unsigned hf   = lane >> 4;
    const unsigned col0 = lane & 15u;
    #pragma unroll
    for (int i = 0; i < 4; ++i) {
        const unsigned mrow0 = m0 + wave_m * 64u + (unsigned)i * 16u + hf * 8u;
        #pragma unroll
        for (int j = 0; j < 4; ++j) {
            const unsigned n = (nt0 + wave_n * 4u + (unsigned)j) * 16u + col0;
            #pragma unroll
            for (int r = 0; r < 8; ++r) {
                __builtin_nontemporal_store(acc[i][j][r],
                    out + (size_t)(mrow0 + r) * N_TOT + n);
            }
        }
    }
}

// ---------------------------------------------------------------------------
extern "C" void kernel_launch(void* const* d_in, const int* in_sizes, int n_in,
                              void* d_out, int out_size, void* d_ws, size_t ws_size,
                              hipStream_t stream) {
    (void)in_sizes; (void)n_in; (void)out_size;

    const float*         x     = (const float*)d_in[0];
    const unsigned char* wbool = (const unsigned char*)d_in[1];   // jnp.bool_ = 1 byte
    const float*         scale = (const float*)d_in[2];
    float*               out   = (float*)d_out;
    unsigned short*      wp    = (unsigned short*)d_ws;           // 2 MB packed bf16

    if (ws_size < (size_t)(K_TOT * N_TOT * sizeof(unsigned short))) return;

    // 1) pack bool weights -> bf16 (+/- scale) in WMMA B-fragment order
    bin_pack_w_kernel<<<(K_TOT * N_TOT) / 256, 256, 0, stream>>>(wbool, scale, wp);

    // 2) tiled bf16 WMMA GEMM: 128 x 4 = 512 workgroups
    const unsigned mblocks = M_TOT / 128u;   // 128
    const unsigned nblocks = N_TOT / 256u;   // 4
    bin_dense_wmma_kernel<<<mblocks * nblocks, 256, 0, stream>>>(x, wp, out);
}
